// GSAPool_55001351192624
// MI455X (gfx1250) — compile-verified, matrix operand order checked by reading, not run on hardware
//
#include <hip/hip_runtime.h>
#include <hip/hip_bf16.h>
#include <math.h>

// Problem geometry (static, from the reference)
#define GB   64                     // graphs
#define GN   2048                   // nodes / graph
#define GD   256                    // feature dim
#define GDEG 16
#define GK   1024                   // kept nodes / graph (ceil(0.5*N))
#define GNT  (GB * GN)              // 131072 nodes
#define GE   (GNT * GDEG)           // 2097152 edges
#define ALPHA_ 0.6f

typedef __attribute__((ext_vector_type(2)))  float    v2f;
typedef __attribute__((ext_vector_type(8)))  float    v8f;
typedef __attribute__((ext_vector_type(16))) _Float16 v16h;

#if defined(__AMDGCN__) && __has_builtin(__builtin_amdgcn_wmma_f32_16x16x4_f32)
#define HAVE_WMMA_F32X4 1
#else
#define HAVE_WMMA_F32X4 0
#endif

// ---------------------------------------------------------------- degree ----
__global__ void k_init_deg(float* __restrict__ deg) {
  int i = blockIdx.x * blockDim.x + threadIdx.x;
  if (i < GNT) deg[i] = 1.0f;                      // self-loop
}

__global__ void k_deg(const int* __restrict__ ei, float* __restrict__ deg) {
  int e = blockIdx.x * blockDim.x + threadIdx.x;
  if (e < GE) atomicAdd(&deg[ei[GE + e]], 1.0f);   // count by target (col)
}

// ------------------------------------------- dual matvec via WMMA (fp32) ----
// One wave per 16-row tile of x.  B is staged in LDS as a zero-padded 16x256
// table (column 0 = gcn_w, column 1 = lin_w, columns 2..15 = 0), so the inner
// loop is pure {global_load_b64 A, ds_load_b64 B, v_wmma} with EXEC untouched
// (WMMA requires EXEC == all 1s).  Accumulator column 0 = x@gcn_w, column 1 =
// x@lin_w, exact fp32.  Two accumulators break the serial WMMA RAW chain.
__global__ void __launch_bounds__(256) k_scores_wmma(
    const float* __restrict__ x, const float* __restrict__ gw,
    const float* __restrict__ lw, float* __restrict__ hg, float* __restrict__ hl)
{
  __shared__ float wtab[16 * GD];                  // 16 KB
  for (int i = threadIdx.x; i < 16 * GD; i += blockDim.x) {
    int col = i >> 8;                              // i / GD
    int k   = i & (GD - 1);
    float v = 0.0f;
    if (col == 0) v = gw[k];
    else if (col == 1) v = lw[k];
    wtab[i] = v;
  }
  __syncthreads();

  const int lane = threadIdx.x & 31;
  const int wave = (blockIdx.x * blockDim.x + threadIdx.x) >> 5;
  const int row0 = wave << 4;                      // first of 16 rows
  const int m    = lane & 15;                      // A: row / B,C: column
  const int hb   = lane >> 4;                      // half-wave selector
  const int koff = hb << 1;                        // this lane's K pair: {0,1} or {2,3}

  // float2 views: element index = (k + koff)/2, both 8-byte aligned
  const v2f* x2 = (const v2f*)(x + (size_t)(row0 + m) * GD + koff);
  const v2f* w2 = (const v2f*)(&wtab[m * GD + koff]);

  v8f c0 = {}, c1 = {};
#if HAVE_WMMA_F32X4
  for (int k0 = 0; k0 < GD; k0 += 8) {             // two K=4 steps per iter
    v2f a0 = x2[(k0 >> 1) + 0];                    // x[row, k0+koff .. +1]
    v2f b0 = w2[(k0 >> 1) + 0];                    // wtab[m, k0+koff .. +1]
    v2f a1 = x2[(k0 >> 1) + 2];                    // x[row, k0+4+koff .. +1]
    v2f b1 = w2[(k0 >> 1) + 2];
    c0 = __builtin_amdgcn_wmma_f32_16x16x4_f32(false, a0, false, b0,
                                               (short)0, c0, false, false);
    c1 = __builtin_amdgcn_wmma_f32_16x16x4_f32(false, a1, false, b1,
                                               (short)0, c1, false, false);
  }
#else
  // Fallback: codegen-confirmed f16 WMMA, same LDS-staged B construction.
  const float* xr = x + (size_t)(row0 + m) * GD;
  const float* wr = &wtab[m * GD];
  for (int k0 = 0; k0 < GD; k0 += 32) {
    v16h a, b;
#pragma unroll
    for (int hi = 0; hi < 16; ++hi) {
      int v  = hi >> 1;
      int kk = ((v & 4) << 2) + ((v & 3) << 1) + (hi & 1) + (hb << 3);
      a[hi] = (_Float16)xr[k0 + kk];
      b[hi] = (_Float16)wr[k0 + kk];
    }
    c0 = __builtin_amdgcn_wmma_f32_16x16x32_f16(false, a, false, b,
                                                (short)0, c0, false, false);
  }
#endif

  // C layout: lane%16 = column, VGPR v = row v (lanes 0-15) / v+8 (lanes 16-31)
  int roff = row0 + (hb << 3);
  if (m == 0) {
#pragma unroll
    for (int v = 0; v < 8; ++v) hg[roff + v] = c0[v] + c1[v];
  } else if (m == 1) {
#pragma unroll
    for (int v = 0; v < 8; ++v) hl[roff + v] = c0[v] + c1[v];
  }
}

// -------------------------------------------------- GCN scalar propagate ----
__global__ void k_selfloop(float* __restrict__ deg_dis,
                           const float* __restrict__ hg, float* __restrict__ acc) {
  int i = blockIdx.x * blockDim.x + threadIdx.x;
  if (i < GNT) {
    float dis = rsqrtf(deg_dis[i]);
    deg_dis[i] = dis;                               // deg -> dis in place
    acc[i] = hg[i] * dis * dis;                     // self-loop contribution
  }
}

__global__ void k_prop(const int* __restrict__ ei, const float* __restrict__ dis,
                       const float* __restrict__ hg, float* __restrict__ acc) {
  int e = blockIdx.x * blockDim.x + threadIdx.x;
  if (e < GE) {
    int s = ei[e], d = ei[GE + e];
    atomicAdd(&acc[d], dis[s] * dis[d] * hg[s]);
  }
}

__global__ void k_score(const float* __restrict__ acc, const float* __restrict__ hl,
                        const float* __restrict__ gb, const float* __restrict__ lb,
                        float* __restrict__ score) {
  int i = blockIdx.x * blockDim.x + threadIdx.x;
  if (i < GNT)
    score[i] = tanhf(ALPHA_ * (acc[i] + gb[0]) + (1.0f - ALPHA_) * (hl[i] + lb[0]));
}

// ------------------------------------------------------- per-graph top-K ----
// rank(i) = #{ j : s[j] > s[i]  or  (s[j] == s[i] and j < i) }  -> total order,
// every rank 0..N-1 hit exactly once; rank < K means kept at slot `rank`.
__global__ void __launch_bounds__(256) k_topk(const float* __restrict__ score,
    int* __restrict__ new_idx, int* __restrict__ perm, int* __restrict__ batch_out)
{
  __shared__ float sc[GN];
  int g = blockIdx.x;
  const float* s = score + (size_t)g * GN;
  for (int i = threadIdx.x; i < GN; i += blockDim.x) sc[i] = s[i];
  __syncthreads();
  const float4* sc4 = (const float4*)sc;
  for (int i = threadIdx.x; i < GN; i += blockDim.x) {
    float si = sc[i];
    int rank = 0;
    for (int j4 = 0; j4 < GN / 4; ++j4) {          // ds_load_b128 per 4 scores
      float4 v = sc4[j4];
      int j = j4 << 2;
      rank += (v.x > si) || (v.x == si && (j + 0) < i);
      rank += (v.y > si) || (v.y == si && (j + 1) < i);
      rank += (v.z > si) || (v.z == si && (j + 2) < i);
      rank += (v.w > si) || (v.w == si && (j + 3) < i);
    }
    int nid = -1;
    if (rank < GK) {
      int pos = g * GK + rank;
      perm[pos]      = g * GN + i;
      batch_out[pos] = g;
      nid = pos;
    }
    new_idx[g * GN + i] = nid;
  }
}

// --------------------------------------------------- gather + scale rows ----
__global__ void k_gather(const float* __restrict__ x, const float* __restrict__ score,
                         const int* __restrict__ perm, float* __restrict__ xout)
{
  int row = blockIdx.x;
  int src = perm[row];
  float sc = score[src];
  const float4* xin = (const float4*)(x + (size_t)src * GD);
  float4* xo = (float4*)(xout + (size_t)row * GD);
  float4 v = xin[threadIdx.x];                     // global_load_b128
  v.x *= sc; v.y *= sc; v.z *= sc; v.w *= sc;
  xo[threadIdx.x] = v;                             // global_store_b128
}

// ------------------------------------------------ edge relabel / filter -----
__global__ void k_edges(const int* __restrict__ ei, const int* __restrict__ new_idx,
                        int* __restrict__ eo, float* __restrict__ keep)
{
  int e = blockIdx.x * blockDim.x + threadIdx.x;
  if (e < GE) {
    int r = new_idx[ei[e]];
    int c = new_idx[ei[GE + e]];
    bool kp = (r >= 0) && (c >= 0);
    eo[e]      = kp ? r : -1;
    eo[GE + e] = kp ? c : -1;
    keep[e]    = kp ? 1.0f : 0.0f;
  }
}

// ---------------------------------------------------------------- launch ----
extern "C" void kernel_launch(void* const* d_in, const int* in_sizes, int n_in,
                              void* d_out, int out_size, void* d_ws, size_t ws_size,
                              hipStream_t stream)
{
  const float* x  = (const float*)d_in[0];
  const int*   ei = (const int*)d_in[1];            // [2, E] row-major
  // d_in[2] = batch (unused: batch_out is the graph id, computed directly)
  const float* gw = (const float*)d_in[3];
  const float* gb = (const float*)d_in[4];
  const float* lw = (const float*)d_in[5];
  const float* lb = (const float*)d_in[6];

  float* ws  = (float*)d_ws;                        // 6*GNT floats = 3 MB
  float* deg = ws;                                  // deg -> dis (in place)
  float* hg  = ws + 1 * GNT;
  float* hl  = ws + 2 * GNT;
  float* acc = ws + 3 * GNT;
  float* scr = ws + 4 * GNT;
  int*   nid = (int*)(ws + 5 * GNT);

  float* out   = (float*)d_out;                     // outputs flat, return order
  float* xout  = out;                               // [B*K, D] f32
  int*   eo    = (int*)(out + (size_t)GB * GK * GD);// [2, E]  i32
  int*   bout  = (int*)(out + (size_t)GB * GK * GD) + 2 * (size_t)GE;  // [B*K]
  int*   permo = bout + GB * GK;                    // [B*K]
  float* keep  = (float*)(permo + GB * GK);         // [E] bool as 0/1

  k_init_deg   <<<GNT / 256, 256, 0, stream>>>(deg);
  k_deg        <<<GE  / 256, 256, 0, stream>>>(ei, deg);
  k_scores_wmma<<<GNT / 16 / 8, 256, 0, stream>>>(x, gw, lw, hg, hl);
  k_selfloop   <<<GNT / 256, 256, 0, stream>>>(deg, hg, acc);
  k_prop       <<<GE  / 256, 256, 0, stream>>>(ei, deg, hg, acc);
  k_score      <<<GNT / 256, 256, 0, stream>>>(acc, hl, gb, lb, scr);
  k_topk       <<<GB, 256, 0, stream>>>(scr, nid, permo, bout);
  k_gather     <<<GB * GK, GD / 4, 0, stream>>>(x, scr, permo, xout);
  k_edges      <<<GE / 256, 256, 0, stream>>>(ei, nid, eo, keep);
}
